// GNNActor_52931176955904
// MI455X (gfx1250) — compile-verified, matrix operand order checked by reading, not run on hardware
//
#include <hip/hip_runtime.h>
#include <hip/hip_bf16.h>

// ---------------------------------------------------------------------------
// GNNActor forward for MI455X (gfx1250, wave32).
// Dense matmuls -> v_wmma_f32_16x16x32_bf16 (bf16 in, f32 accumulate).
// All tile loads are unconditional (row-clamp + zero-padded LDS weights),
// so they lower to global_load_b128 / ds_load_b128 + v_cvt_pk_bf16_f32.
// Edge aggregation -> wave-per-edge global_atomic_add_f32 (bandwidth bound).
// ---------------------------------------------------------------------------

typedef __attribute__((ext_vector_type(16))) __bf16 v16bf;
typedef __attribute__((ext_vector_type(8)))  float  v8f;

__device__ __forceinline__ float leakyf(float v) { return v > 0.0f ? v : 0.01f * v; }

__device__ __forceinline__ v8f wmma_bf16(v16bf a, v16bf b, v8f c) {
  return __builtin_amdgcn_wmma_f32_16x16x32_bf16(false, a, false, b, (short)0, c,
                                                 false, false);
}

// Unconditional 16x32 tile load (f32 -> bf16), CDNA5 WMMA A/B layout.
// element i of lane: k = kbase + half*8 + (i<8 ? i : 8+i), half = lane>>4.
// Per lane this is two contiguous 8-float (32B) runs -> global_load_b128 x2.
__device__ __forceinline__ v16bf load_tile_g(const float* __restrict__ base, int ld,
                                             int row, int kbase, int half) {
  const float* p = base + (long long)row * ld + kbase + half * 8;
  v16bf t;
#pragma unroll
  for (int i = 0; i < 8; ++i) t[i] = (__bf16)p[i];
#pragma unroll
  for (int i = 0; i < 8; ++i) t[8 + i] = (__bf16)p[16 + i];
  return t;
}

// Same layout from pre-padded bf16 LDS weights -> ds_load_b128 x2.
__device__ __forceinline__ v16bf load_tile_lds(const __bf16* base, int ld,
                                               int row, int kbase, int half) {
  const __bf16* p = base + row * ld + kbase + half * 8;
  v16bf t;
#pragma unroll
  for (int i = 0; i < 8; ++i) t[i] = p[i];
#pragma unroll
  for (int i = 0; i < 8; ++i) t[8 + i] = p[16 + i];
  return t;
}

// ---------------------------------------------------------------------------

__global__ void k_zero(float* __restrict__ p, long long n) {
  long long i = (long long)blockIdx.x * blockDim.x + threadIdx.x;
  const long long stride = (long long)gridDim.x * blockDim.x;
  for (; i < n; i += stride) p[i] = 0.0f;
}

// deg[dst] += 1 per edge (edge_index int32: [0..E) = src, [E..2E) = dst)
__global__ void k_degree(const int* __restrict__ ei, float* __restrict__ deg, int E) {
  const int e = blockIdx.x * blockDim.x + threadIdx.x;
  if (e < E) atomicAdd(&deg[ei[E + e]], 1.0f);
}

// dinv = rsqrt(deg + 1)   (self loop included)
__global__ void k_dinv(float* __restrict__ deg, int n) {
  const int i = blockIdx.x * blockDim.x + threadIdx.x;
  if (i < n) deg[i] = rsqrtf(deg[i] + 1.0f);
}

// h = state @ convW^T    [N,64] = [N,64] x [64,64]^T     (WMMA; N % 16 == 0)
__global__ void k_conv_gemm(const float* __restrict__ state,
                            const float* __restrict__ convW,
                            float* __restrict__ h, int nrows) {
  const int wave = (blockIdx.x * blockDim.x + threadIdx.x) >> 5;
  const int lane = threadIdx.x & 31;
  const int mtiles = nrows >> 4;
  if (wave >= mtiles * 4) return;
  const int row0 = (wave >> 2) << 4;
  const int col0 = (wave & 3) << 4;
  const int r    = row0 + (lane & 15);
  const int rw   = col0 + (lane & 15);
  const int half = lane >> 4;
  v8f acc = {};
#pragma unroll
  for (int kt = 0; kt < 2; ++kt) {
    v16bf a = load_tile_g(state, 64, r,  kt * 32, half);
    v16bf b = load_tile_g(convW, 64, rw, kt * 32, half);
    acc = wmma_bf16(a, b, acc);
  }
  const int n = col0 + (lane & 15);
  const int rbase = row0 + half * 8;
#pragma unroll
  for (int j = 0; j < 8; ++j)
    h[(long long)(rbase + j) * 64 + n] = acc[j];
}

// agg[dst] += h[src] * dinv[src]*dinv[dst]  -- one wave per edge, 2 ch/lane
__global__ void k_scatter(const int* __restrict__ ei, const float* __restrict__ h,
                          const float* __restrict__ dinv, float* __restrict__ agg,
                          int E) {
  const int e = (int)((blockIdx.x * (long long)blockDim.x + threadIdx.x) >> 5);
  if (e >= E) return;
  const int lane = threadIdx.x & 31;
  const int s = ei[e];
  const int d = ei[E + e];
  const float norm = dinv[s] * dinv[d];
  const int c = lane * 2;
  const long long sb = (long long)s * 64 + c;
  const long long db = (long long)d * 64 + c;
  atomicAdd(&agg[db],     h[sb]     * norm);
  atomicAdd(&agg[db + 1], h[sb + 1] * norm);
}

// x = relu(agg + h*dinv^2 + convb) + state   (written over h)
__global__ void k_selfbias(const float* __restrict__ agg, float* __restrict__ h,
                           const float* __restrict__ dinv,
                           const float* __restrict__ convb,
                           const float* __restrict__ state, long long total) {
  const long long i = (long long)blockIdx.x * blockDim.x + threadIdx.x;
  if (i >= total) return;
  const int n = (int)(i >> 6);
  const int c = (int)(i & 63);
  const float di = dinv[n];
  const float v = agg[i] + h[i] * di * di + convb[c];
  h[i] = fmaxf(v, 0.0f) + state[i];
}

// xg1 = leaky_relu(state_flat[B,384] @ gW1^T + gb1)   [B,256]   (WMMA)
__global__ void k_glob1(const float* __restrict__ state, const float* __restrict__ gW1,
                        const float* __restrict__ gb1, float* __restrict__ xg1, int Bn) {
  const int wave = (blockIdx.x * blockDim.x + threadIdx.x) >> 5;
  const int lane = threadIdx.x & 31;
  const int mtiles = (Bn + 15) >> 4;
  if (wave >= mtiles * 16) return;
  const int row0 = (wave >> 4) << 4;
  const int col0 = (wave & 15) << 4;
  const int half = lane >> 4;
  int r = row0 + (lane & 15);
  r = r < Bn ? r : Bn - 1;                  // clamp: load valid rows, mask at store
  const int rw = col0 + (lane & 15);
  v8f acc = {};
  for (int kt = 0; kt < 12; ++kt) {
    v16bf a = load_tile_g(state, 384, r,  kt * 32, half);
    v16bf b = load_tile_g(gW1,   384, rw, kt * 32, half);
    acc = wmma_bf16(a, b, acc);
  }
  const int n = col0 + (lane & 15);
  const int rbase = row0 + half * 8;
#pragma unroll
  for (int j = 0; j < 8; ++j) {
    const int rr = rbase + j;
    if (rr < Bn) xg1[(long long)rr * 256 + n] = leakyf(acc[j] + gb1[n]);
  }
}

// xg2p = leaky_relu(xg1 @ gW2^T + gb2), stored padded [B,16] (cols 10..15 = 0)
__global__ void k_glob2(const float* __restrict__ xg1, const float* __restrict__ gW2,
                        const float* __restrict__ gb2, float* __restrict__ xg2p,
                        int Bn) {
  __shared__ __align__(16) __bf16 sW[16 * 256];   // gW2 zero-padded to 16 rows
  for (int idx = threadIdx.x; idx < 16 * 256; idx += blockDim.x) {
    const int n = idx >> 8;
    const int k = idx & 255;
    sW[idx] = (__bf16)(n < 10 ? gW2[n * 256 + k] : 0.0f);
  }
  __syncthreads();

  const int wave = (blockIdx.x * blockDim.x + threadIdx.x) >> 5;
  const int lane = threadIdx.x & 31;
  const int mtiles = (Bn + 15) >> 4;
  if (wave >= mtiles) return;
  const int row0 = wave << 4;
  const int half = lane >> 4;
  int r = row0 + (lane & 15);
  r = r < Bn ? r : Bn - 1;
  const int rw = lane & 15;
  v8f acc = {};
#pragma unroll
  for (int kt = 0; kt < 8; ++kt) {
    v16bf a = load_tile_g(xg1, 256, r, kt * 32, half);
    v16bf b = load_tile_lds(sW, 256, rw, kt * 32, half);
    acc = wmma_bf16(a, b, acc);
  }
  const int n = lane & 15;
  const int rbase = row0 + half * 8;
#pragma unroll
  for (int j = 0; j < 8; ++j) {
    const int rr = rbase + j;
    if (rr < Bn)
      xg2p[(long long)rr * 16 + n] = n < 10 ? leakyf(acc[j] + gb2[n]) : 0.0f;
  }
}

// x1 = leaky_relu(concat(x, xg2)[N,74] @ l1W^T + l1b)  [N,32]  (WMMA; N%16==0)
// K padded 74 -> 96; l1W staged zero-padded in LDS; xg2p already padded.
__global__ void k_l1(const float* __restrict__ X, const float* __restrict__ xg2p,
                     const float* __restrict__ l1W, const float* __restrict__ l1b,
                     float* __restrict__ x1, int nrows) {
  __shared__ __align__(16) __bf16 sW[32 * 96];    // l1W zero-padded K: 74 -> 96
  for (int idx = threadIdx.x; idx < 32 * 96; idx += blockDim.x) {
    const int n = idx / 96;
    const int k = idx - n * 96;
    sW[idx] = (__bf16)(k < 74 ? l1W[n * 74 + k] : 0.0f);
  }
  __syncthreads();

  const int wave = (blockIdx.x * blockDim.x + threadIdx.x) >> 5;
  const int lane = threadIdx.x & 31;
  const int mtiles = nrows >> 4;
  if (wave >= mtiles * 2) return;
  const int row0 = (wave >> 1) << 4;
  const int col0 = (wave & 1) << 4;
  const int r    = row0 + (lane & 15);
  const int rw   = col0 + (lane & 15);
  const int half = lane >> 4;
  v8f acc = {};
#pragma unroll
  for (int kt = 0; kt < 2; ++kt) {           // k 0..63 from X
    v16bf a = load_tile_g(X, 64, r, kt * 32, half);
    v16bf b = load_tile_lds(sW, 96, rw, kt * 32, half);
    acc = wmma_bf16(a, b, acc);
  }
  {                                          // k 64..95: run1 from xg2p, run2 zero
    const float* q = xg2p + (long long)(r / 6) * 16 + half * 8;
    v16bf a;
#pragma unroll
    for (int i = 0; i < 8; ++i) a[i] = (__bf16)q[i];
#pragma unroll
    for (int i = 0; i < 8; ++i) a[8 + i] = (__bf16)0.0f;
    v16bf b = load_tile_lds(sW, 96, rw, 64, half);
    acc = wmma_bf16(a, b, acc);
  }
  const int n = col0 + (lane & 15);
  const int rbase = row0 + half * 8;
#pragma unroll
  for (int j = 0; j < 8; ++j)
    x1[(long long)(rbase + j) * 32 + n] = leakyf(acc[j] + l1b[n]);
}

// x2 = leaky_relu(x1 @ l2W^T + l2b)   [N,32]   (WMMA, single K tile; N%16==0)
__global__ void k_l2(const float* __restrict__ x1, const float* __restrict__ l2W,
                     const float* __restrict__ l2b, float* __restrict__ x2, int nrows) {
  const int wave = (blockIdx.x * blockDim.x + threadIdx.x) >> 5;
  const int lane = threadIdx.x & 31;
  const int mtiles = nrows >> 4;
  if (wave >= mtiles * 2) return;
  const int row0 = (wave >> 1) << 4;
  const int col0 = (wave & 1) << 4;
  const int r    = row0 + (lane & 15);
  const int rw   = col0 + (lane & 15);
  const int half = lane >> 4;
  v16bf a = load_tile_g(x1,  32, r,  0, half);
  v16bf b = load_tile_g(l2W, 32, rw, 0, half);
  v8f acc = {};
  acc = wmma_bf16(a, b, acc);
  const int n = col0 + (lane & 15);
  const int rbase = row0 + half * 8;
#pragma unroll
  for (int j = 0; j < 8; ++j)
    x2[(long long)(rbase + j) * 32 + n] = leakyf(acc[j] + l2b[n]);
}

// conc = softplus(x2 @ l3W^T + l3b); block-reduce sum(|conc|) -> sumabs
__global__ void k_l3(const float* __restrict__ x2, const float* __restrict__ l3W,
                     const float* __restrict__ l3b, float* __restrict__ conc,
                     float* __restrict__ sumabs, int nrows) {
  const int n = blockIdx.x * blockDim.x + threadIdx.x;
  float c = 0.0f;
  if (n < nrows) {
    float s = l3b[0];
#pragma unroll
    for (int k = 0; k < 32; ++k) s += x2[(long long)n * 32 + k] * l3W[k];
    // numerically stable softplus (result >= 0, so |conc| == conc)
    c = fmaxf(s, 0.0f) + log1pf(expf(-fabsf(s)));
    conc[n] = c;
  }
  __shared__ float red[256];
  red[threadIdx.x] = c;
  __syncthreads();
  for (int s = 128; s > 0; s >>= 1) {
    if (threadIdx.x < s) red[threadIdx.x] += red[threadIdx.x + s];
    __syncthreads();
  }
  if (threadIdx.x == 0) atomicAdd(sumabs, red[0]);
}

// action[b,a] = conc[b,a] / (sum_a conc + 1e-20)
__global__ void k_action(const float* __restrict__ conc, float* __restrict__ out, int Bn) {
  const int b = blockIdx.x * blockDim.x + threadIdx.x;
  if (b >= Bn) return;
  float v[6];
  float s = 1e-20f;
#pragma unroll
  for (int a = 0; a < 6; ++a) { v[a] = conc[b * 6 + a]; s += v[a]; }
  const float inv = 1.0f / s;
#pragma unroll
  for (int a = 0; a < 6; ++a) out[b * 6 + a] = v[a] * inv;
}

__global__ void k_reg(const float* __restrict__ sumabs, float* __restrict__ out,
                      int nrows) {
  if (threadIdx.x == 0 && blockIdx.x == 0)
    out[nrows] = sumabs[0] / (float)nrows;
}

// ---------------------------------------------------------------------------

extern "C" void kernel_launch(void* const* d_in, const int* in_sizes, int n_in,
                              void* d_out, int out_size, void* d_ws, size_t ws_size,
                              hipStream_t stream) {
  const float* state = (const float*)d_in[0];
  const int*   ei    = (const int*)  d_in[1];   // int32 edge_index [2,E]
  const float* convW = (const float*)d_in[2];
  const float* convb = (const float*)d_in[3];
  const float* gW1   = (const float*)d_in[4];
  const float* gb1   = (const float*)d_in[5];
  const float* gW2   = (const float*)d_in[6];
  const float* gb2   = (const float*)d_in[7];
  const float* l1W   = (const float*)d_in[8];
  const float* l1b   = (const float*)d_in[9];
  const float* l2W   = (const float*)d_in[10];
  const float* l2b   = (const float*)d_in[11];
  const float* l3W   = (const float*)d_in[12];
  const float* l3b   = (const float*)d_in[13];

  const int Nn = in_sizes[0] / 64;   // 150000
  const int E  = in_sizes[1] / 2;    // 1200000
  const int Bn = Nn / 6;             // 25000

  float* ws = (float*)d_ws;
  size_t o = 0;
  auto take = [&](size_t n) { float* p = ws + o; o += (n + 255) & ~(size_t)255; return p; };
  float* deg  = take((size_t)Nn);          // -> dinv in place
  float* h    = take((size_t)Nn * 64);     // -> x (residual) in place
  float* agg  = take((size_t)Nn * 64);     // reused as xg1 after k_selfbias
  float* xg1  = agg;                       // Bn*256 <= Nn*64
  float* xg2p = take((size_t)Bn * 16);     // padded [B,16], cols 10..15 zero
  float* x1   = take((size_t)Nn * 32);
  float* x2   = take((size_t)Nn * 32);
  float* conc = take((size_t)Nn);
  float* sum1 = take((size_t)1);
  (void)ws_size; (void)n_in; (void)out_size;

  float* out = (float*)d_out;

  const int T = 256;                       // 8 waves / block
  auto waves_to_blocks = [&](long long w) { return (int)((w * 32 + T - 1) / T); };

  // zero accumulators
  k_zero<<<1024, T, 0, stream>>>(deg, (long long)Nn);
  k_zero<<<4096, T, 0, stream>>>(agg, (long long)Nn * 64);
  k_zero<<<1, 32, 0, stream>>>(sum1, 1);

  // GCN conv
  k_degree<<<(E + T - 1) / T, T, 0, stream>>>(ei, deg, E);
  k_dinv<<<(Nn + T - 1) / T, T, 0, stream>>>(deg, Nn);
  {
    const long long waves = (long long)(Nn / 16) * 4;
    k_conv_gemm<<<waves_to_blocks(waves), T, 0, stream>>>(state, convW, h, Nn);
  }
  k_scatter<<<waves_to_blocks(E), T, 0, stream>>>(ei, h, deg, agg, E);
  {
    const long long total = (long long)Nn * 64;
    k_selfbias<<<(int)((total + T - 1) / T), T, 0, stream>>>(agg, h, deg, convb, state, total);
  }

  // global branch (runs after k_selfbias because xg1 aliases agg)
  {
    const long long waves = (long long)((Bn + 15) / 16) * 16;
    k_glob1<<<waves_to_blocks(waves), T, 0, stream>>>(state, gW1, gb1, xg1, Bn);
  }
  {
    const long long waves = (long long)((Bn + 15) / 16);
    k_glob2<<<waves_to_blocks(waves), T, 0, stream>>>(xg1, gW2, gb2, xg2p, Bn);
  }

  // per-node head
  {
    const long long waves = (long long)(Nn / 16) * 2;
    k_l1<<<waves_to_blocks(waves), T, 0, stream>>>(h, xg2p, l1W, l1b, x1, Nn);
    k_l2<<<waves_to_blocks(waves), T, 0, stream>>>(x1, l2W, l2b, x2, Nn);
  }
  k_l3<<<(Nn + T - 1) / T, T, 0, stream>>>(x2, l3W, l3b, conc, sum1, Nn);

  // outputs
  k_action<<<(Bn + T - 1) / T, T, 0, stream>>>(conc, out, Bn);
  k_reg<<<1, 32, 0, stream>>>(sum1, out, Nn);
}
